// EulerIntegrator_72593537237650
// MI455X (gfx1250) — matrix operand order repaired and checked by reading, not compiled, not run on hardware
//
#include <hip/hip_runtime.h>

// ---------------------------------------------------------------------------
// EulerIntegrator on MI455X (gfx1250): bf16 WMMA path, f32 state resident in LDS
//   B=4096 rows, D=1024, R=256, steps read from device scalar.
//   One block = 16 rows; all steps integrated in-block.
//   v-state: f32 master copy in LDS.  x-state: accumulated in out_x (global).
// ---------------------------------------------------------------------------

typedef __attribute__((ext_vector_type(16))) __bf16 v16bf;
typedef __attribute__((ext_vector_type(8)))  float  v8f;

#define DVAL 1024
#define RVAL 256
#define DT_STEP 0.01f

__device__ __forceinline__ unsigned short f2bf(float x) {
    unsigned u = __builtin_bit_cast(unsigned, x);
    u += 0x7FFFu + ((u >> 16) & 1u);            // round-to-nearest-even
    return (unsigned short)(u >> 16);
}

union FragCast { uint4 q[2]; v16bf v; };

__device__ __forceinline__ v16bf load_frag(const unsigned short* p) {
    FragCast u;
    u.q[0] = *(const uint4*)(p);
    u.q[1] = *(const uint4*)(p + 8);
    return u.v;
}

// Gather a WMMA B-fragment (32x16, bf16) straight from a row-major f32 matrix.
// B layout: lane = 16*hi + n; element pair j holds K = 32*kt + 16*hi + 2j, +1.
__device__ __forceinline__ v16bf gather_b(const float* __restrict__ M, int ld,
                                          int kt, int nt, int hi, int n) {
    const int col = nt * 16 + n;
    const int kb  = kt * 32 + hi * 16;
    unsigned w[8];
#pragma unroll
    for (int j = 0; j < 8; ++j) {
        float a = M[(kb + 2 * j)     * ld + col];
        float b = M[(kb + 2 * j + 1) * ld + col];
        w[j] = (unsigned)f2bf(a) | ((unsigned)f2bf(b) << 16);
    }
    FragCast u;
    u.q[0] = make_uint4(w[0], w[1], w[2], w[3]);
    u.q[1] = make_uint4(w[4], w[5], w[6], w[7]);
    return u.v;
}

// ---------------------------------------------------------------------------
// Pack kernels: pre-swizzle U (D x R) and W (R x D) into bf16 WMMA B-fragments.
// Fragment = 1024 bytes: lane-major, 16 bf16 per lane (two b128 loads at use).
// ---------------------------------------------------------------------------
__global__ __launch_bounds__(256) void pack_b_frags(const float* __restrict__ M,
                                                    unsigned short* __restrict__ out,
                                                    int ld, int kt_per_col) {
    const int t    = blockIdx.x * 256 + threadIdx.x;
    const int lane = t & 31;
    const int frag = t >> 5;                    // frag = nt*kt_per_col + kt
    const int kt   = frag % kt_per_col;
    const int nt   = frag / kt_per_col;
    const int hi   = lane >> 4;
    const int n    = lane & 15;
    const int col  = nt * 16 + n;
    const int kb   = kt * 32 + hi * 16;
    unsigned w[8];
#pragma unroll
    for (int j = 0; j < 8; ++j) {
        float a = M[(kb + 2 * j)     * ld + col];
        float b = M[(kb + 2 * j + 1) * ld + col];
        w[j] = (unsigned)f2bf(a) | ((unsigned)f2bf(b) << 16);
    }
    unsigned short* o = out + frag * 512 + lane * 16;
    *(uint4*)(o)     = make_uint4(w[0], w[1], w[2], w[3]);
    *(uint4*)(o + 8) = make_uint4(w[4], w[5], w[6], w[7]);
}

// ---------------------------------------------------------------------------
// Software-pipelined K-loop: prefetch next A/B fragments before each WMMA.
// ---------------------------------------------------------------------------
template <bool PACKED>
__device__ __forceinline__ v8f gemm_tile(const unsigned short* __restrict__ aBase,
                                         const unsigned short* __restrict__ bBase,
                                         const float* __restrict__ Mfall, int ld,
                                         int nt, int nkt, int hi, int ml) {
    v8f acc = {0.f, 0.f, 0.f, 0.f, 0.f, 0.f, 0.f, 0.f};
    v16bf a = load_frag(aBase);
    v16bf b = PACKED ? load_frag(bBase) : gather_b(Mfall, ld, 0, nt, hi, ml);
#pragma unroll 2
    for (int kt = 0; kt < nkt - 1; ++kt) {
        v16bf an = load_frag(aBase + (kt + 1) * 512);
        v16bf bn = PACKED ? load_frag(bBase + (kt + 1) * 512)
                          : gather_b(Mfall, ld, kt + 1, nt, hi, ml);
        acc = __builtin_amdgcn_wmma_f32_16x16x32_bf16(
            false, a, false, b, (short)0, acc, false, false);
        a = an; b = bn;
    }
    acc = __builtin_amdgcn_wmma_f32_16x16x32_bf16(
        false, a, false, b, (short)0, acc, false, false);
    return acc;
}

// ---------------------------------------------------------------------------
// Main persistent-slab integrator. 256 threads = 8 wave32s, 16 rows per block.
// Dynamic LDS: vF (16x1024 f32, 64K) | vA (A-frags bf16, 32K) |
//              p2 (16x256 bf16 row-major, 8K) | p2A (A-frags, 8K) = 112K.
// ---------------------------------------------------------------------------
template <bool PACKED>
__global__ __launch_bounds__(256) void integrate_kernel(
    const float* __restrict__ x0, const float* __restrict__ v0,
    const float* __restrict__ force,
    const float* __restrict__ U, const float* __restrict__ W,
    const unsigned short* __restrict__ Uf, const unsigned short* __restrict__ Wf,
    const int* __restrict__ stepsPtr,
    float* __restrict__ outX, float* __restrict__ outV) {

    extern __shared__ char smem[];
    float*          vF  = (float*)(smem);                    // 65536 B
    unsigned short* vA  = (unsigned short*)(smem + 65536);   // 32768 B
    unsigned short* p2  = (unsigned short*)(smem + 98304);   //  8192 B
    unsigned short* p2A = (unsigned short*)(smem + 106496);  //  8192 B

    const int tid  = threadIdx.x;
    const int lane = tid & 31;
    const int wave = tid >> 5;
    const int hi   = lane >> 4;
    const int ml   = lane & 15;
    const long base = (long)blockIdx.x * 16 * DVAL;
    const int steps = *stepsPtr;
    const float dt = DT_STEP;

    // prologue: load v slab into LDS, seed out_x with x0
#pragma unroll 4
    for (int i = 0; i < 64; ++i) {
        const int e = tid + 256 * i;
        vF[e] = v0[base + e];
        outX[base + e] = x0[base + e];
    }
    __syncthreads();

    for (int s = 0; s < steps; ++s) {
        // --- phase 1: x += dt * v (pre-update v), build bf16 A-fragments of v
#pragma unroll 4
        for (int i = 0; i < 64; ++i) {
            const int e = tid + 256 * i;
            outX[base + e] += dt * vF[e];
        }
#pragma unroll 1
        for (int c = tid; c < 1024; c += 256) {      // 32 ktiles * 32 lanes
            const int kt = c >> 5;
            const int ln = c & 31;
            const int m  = ln & 15;
            const int h8 = (ln >> 4) * 16;
            const float* vr = vF + m * DVAL + kt * 32 + h8;
            unsigned w[8];
#pragma unroll
            for (int j = 0; j < 8; ++j) {
                const int off = 2 * j;  // A-frag K layout
                w[j] = (unsigned)f2bf(vr[off]) | ((unsigned)f2bf(vr[off + 1]) << 16);
            }
            unsigned short* o = vA + c * 16;
            *(uint4*)(o)     = make_uint4(w[0], w[1], w[2], w[3]);
            *(uint4*)(o + 8) = make_uint4(w[4], w[5], w[6], w[7]);
        }
        __syncthreads();

        // --- phase 2: p = v @ U  (M=16, N=256, K=1024), then p^2 -> LDS
#pragma unroll 1
        for (int i = 0; i < 2; ++i) {
            const int nt = wave + 8 * i;             // 16 N-tiles over R
            v8f acc = gemm_tile<PACKED>(vA + lane * 16,
                                        Uf + nt * 32 * 512 + lane * 16,
                                        U, RVAL, nt, 32, hi, ml);
            const int col = nt * 16 + ml;
#pragma unroll
            for (int r = 0; r < 8; ++r) {
                const float p = acc[r];
                p2[(r + 8 * hi) * RVAL + col] = f2bf(p * p);
            }
        }
        __syncthreads();

        // --- phase 3: repack p^2 into A-fragment order (8 ktiles * 32 lanes)
        {
            const int kt = tid >> 5;
            const int ln = tid & 31;
            const int m  = ln & 15;
            const int h8 = (ln >> 4) * 16;
            const unsigned* pr = (const unsigned*)(p2 + m * RVAL + kt * 32 + h8);
            unsigned w[8];
#pragma unroll
            for (int j = 0; j < 8; ++j) w[j] = pr[j];
            unsigned short* o = p2A + tid * 16;
            *(uint4*)(o)     = make_uint4(w[0], w[1], w[2], w[3]);
            *(uint4*)(o + 8) = make_uint4(w[4], w[5], w[6], w[7]);
        }
        __syncthreads();

        // --- phase 4: g = (p^2) @ W (M=16, N=1024, K=256); v += dt*(f - g)
#pragma unroll 1
        for (int i = 0; i < 8; ++i) {
            const int nt = wave + 8 * i;             // 64 N-tiles over D
            v8f acc = gemm_tile<PACKED>(p2A + lane * 16,
                                        Wf + nt * 8 * 512 + lane * 16,
                                        W, DVAL, nt, 8, hi, ml);
            const int col = nt * 16 + ml;
#pragma unroll
            for (int r = 0; r < 8; ++r) {
                const int m = r + 8 * hi;
                const float fo = force[base + m * DVAL + col];
                vF[m * DVAL + col] += dt * (fo - acc[r]);
            }
        }
        __syncthreads();
    }

    // --- epilogue: v = final state (x already accumulated in out_x)
#pragma unroll 4
    for (int i = 0; i < 64; ++i) {
        const int e = tid + 256 * i;
        outV[base + e] = vF[e];
    }
}

// ---------------------------------------------------------------------------
extern "C" void kernel_launch(void* const* d_in, const int* in_sizes, int n_in,
                              void* d_out, int out_size, void* d_ws, size_t ws_size,
                              hipStream_t stream) {
    const float* x     = (const float*)d_in[0];
    const float* v     = (const float*)d_in[1];
    const float* force = (const float*)d_in[2];
    const float* U     = (const float*)d_in[3];
    const float* W     = (const float*)d_in[4];
    const int*   steps = (const int*)d_in[5];

    const int rows   = in_sizes[0] / DVAL;     // 4096
    const int blocks = rows / 16;              // 256
    float* outX = (float*)d_out;
    float* outV = outX + (long)rows * DVAL;

    const size_t fragBytes = 512u * 1024u;     // 512 fragments * 1KB each side
    const bool packed = ws_size >= 2 * fragBytes;

    const int ldsBytes = 112 * 1024;

    if (packed) {
        unsigned short* Uf = (unsigned short*)d_ws;
        unsigned short* Wf = Uf + 512 * 512;
        // U: D x R -> 16 ntiles * 32 ktiles; W: R x D -> 64 ntiles * 8 ktiles
        pack_b_frags<<<64, 256, 0, stream>>>(U, Uf, RVAL, 32);
        pack_b_frags<<<64, 256, 0, stream>>>(W, Wf, DVAL, 8);

        hipFuncSetAttribute(reinterpret_cast<const void*>(integrate_kernel<true>),
                            hipFuncAttributeMaxDynamicSharedMemorySize, ldsBytes);
        integrate_kernel<true><<<blocks, 256, ldsBytes, stream>>>(
            x, v, force, U, W, Uf, Wf, steps, outX, outV);
    } else {
        hipFuncSetAttribute(reinterpret_cast<const void*>(integrate_kernel<false>),
                            hipFuncAttributeMaxDynamicSharedMemorySize, ldsBytes);
        integrate_kernel<false><<<blocks, 256, ldsBytes, stream>>>(
            x, v, force, U, W, nullptr, nullptr, steps, outX, outV);
    }
}
